// SelfAttention_31318901522889
// MI455X (gfx1250) — matrix operand order repaired and checked by reading, not compiled, not run on hardware
//
#include <hip/hip_runtime.h>

#define BB 4
#define SS 2048
#define DD 256
#define NHD 12
#define FF (NHD*DD)   // 3072 output features per projection

typedef unsigned short bf16_t;
typedef __attribute__((ext_vector_type(16))) __bf16 v16bf;
typedef __attribute__((ext_vector_type(8)))  float  v8f;

union BV16 { uint4 q[2]; v16bf v; unsigned short h[16]; };

__device__ __forceinline__ unsigned short f2bf(float f) {
    unsigned u = __float_as_uint(f);
    u += 0x7FFFu + ((u >> 16) & 1u);   // round-to-nearest-even
    return (unsigned short)(u >> 16);
}

__device__ __forceinline__ v8f v8f_zero() {
    v8f z;
    #pragma unroll
    for (int i = 0; i < 8; ++i) z[i] = 0.0f;
    return z;
}

__device__ __forceinline__ v8f wmma_bf16(v16bf a, v16bf b, v8f c) {
    // D = A(16x32 bf16) * B(32x16 bf16) + C(16x16 f32)
    return __builtin_amdgcn_wmma_f32_16x16x32_bf16(false, a, false, b, (short)0, c,
                                                   false, false);
}

// async 16B copy Global -> LDS, tracked by ASYNCcnt (CDNA5)
__device__ __forceinline__ void async_cp16(unsigned lds_off, const void* gaddr) {
    asm volatile("global_load_async_to_lds_b128 %0, %1, off"
                 :: "v"(lds_off), "v"(gaddr) : "memory");
}
__device__ __forceinline__ void wait_async0() {
    asm volatile("s_wait_asynccnt 0" ::: "memory");
}

// ---------------- prep kernels ----------------
__global__ void zero_f32(float* p, int n) {
    int i = blockIdx.x * blockDim.x + threadIdx.x;
    if (i < n) p[i] = 0.0f;
}

__global__ void cvt_bf16(const float* __restrict__ in, bf16_t* __restrict__ out, int n) {
    int i = blockIdx.x * blockDim.x + threadIdx.x;
    if (i < n) out[i] = f2bf(in[i]);
}

// W [C, N, D] f32 -> Wt [N*D, C] bf16 (row = output feature, contiguous over c)
__global__ void transpose_w(const float* __restrict__ w, bf16_t* __restrict__ wt) {
    int i = blockIdx.x * blockDim.x + threadIdx.x;   // over FF*DD
    if (i >= FF * DD) return;
    int c    = i & (DD - 1);
    int feat = i >> 8;                 // n*DD + d
    int n = feat >> 8;
    int d = feat & (DD - 1);
    wt[i] = f2bf(w[((size_t)c * NHD + n) * DD + d]);
}

// ---------------- projection GEMM: [S,256] x [256, 3072] per batch ----------------
// MODE 0: store rows  -> Kp [B,N,S,D]   (bf16)
// MODE 1: store trans -> Vt [B,N,D,S]   (bf16)
template <int MODE>
__global__ __launch_bounds__(32) void proj_kernel(
    const bf16_t* __restrict__ X,     // [B,S,D] bf16
    const bf16_t* __restrict__ Wt,    // [FF, D] bf16
    const float*  __restrict__ bias,  // [FF]
    bf16_t* __restrict__ outp)
{
    const int lane = threadIdx.x & 31;
    const int col  = lane & 15;
    const int hi   = lane >> 4;

    int blk = blockIdx.x;                       // B * (SS/16) * (FF/16)
    const int FT = FF / 16;                     // 192
    int ft = blk % FT;        blk /= FT;
    int st = blk % (SS / 16); blk /= (SS / 16);
    int b  = blk;
    const int feat = ft * 16 + col;             // B row / C column
    const int s0   = st * 16;

    const bf16_t* arow = X  + ((size_t)b * SS + s0 + col) * DD;  // A row (s = col)
    const bf16_t* brow = Wt + (size_t)feat * DD;                 // B "column" (contig)

    v8f acc = v8f_zero();
    #pragma unroll
    for (int c0 = 0; c0 < DD; c0 += 32) {
        BV16 a, bm;
        a.q[0]  = *(const uint4*)(arow + c0 + hi * 8);
        a.q[1]  = *(const uint4*)(arow + c0 + hi * 8 + 16);
        bm.q[0] = *(const uint4*)(brow + c0 + hi * 8);
        bm.q[1] = *(const uint4*)(brow + c0 + hi * 8 + 16);
        acc = wmma_bf16(a.v, bm.v, acc);
    }
    const float bsc = bias[feat];
    const int n = feat >> 8;
    const int d = feat & (DD - 1);
    if (MODE == 0) {
        bf16_t* dst = outp + (((size_t)b * NHD + n) * SS) * DD + d;
        #pragma unroll
        for (int r = 0; r < 8; ++r)
            dst[(size_t)(s0 + r + 8 * hi) * DD] = f2bf(acc[r] + bsc);
    } else {
        // lane holds 8 consecutive s for fixed d -> one 16B store into Vt row d
        unsigned pk[4];
        #pragma unroll
        for (int r = 0; r < 4; ++r) {
            unsigned lo = f2bf(acc[2 * r] + bsc);
            unsigned hh = f2bf(acc[2 * r + 1] + bsc);
            pk[r] = lo | (hh << 16);
        }
        uint4 q; q.x = pk[0]; q.y = pk[1]; q.z = pk[2]; q.w = pk[3];
        *(uint4*)(outp + (((size_t)b * NHD + n) * DD + d) * (size_t)SS + s0 + 8 * hi) = q;
    }
}

// ---------------- fused flash-attention (transposed-scores, async LDS staging) ----
__global__ __launch_bounds__(256) void attn_kernel(
    const bf16_t* __restrict__ X,    // [B,S,D]    queries (shared over heads)
    const bf16_t* __restrict__ Kp,   // [B,N,S,D]  keys   (qk projection)
    const bf16_t* __restrict__ Vt,   // [B,N,D,S]  values, d-major
    const int*    __restrict__ mask, // [B,S,S]
    float* __restrict__ out)         // [B,S,D], accumulated over heads
{
    // double-buffered workgroup tiles: K tile 32f x 256d, V tile 256d x 32f
    __shared__ __align__(16) bf16_t Ks[2][32 * DD];
    __shared__ __align__(16) bf16_t Vs[2][DD * 32];

    const int tid  = threadIdx.x;
    const int lane = tid & 31;
    const int wave = tid >> 5;
    const int col  = lane & 15;
    const int hi   = lane >> 4;

    int blk = blockIdx.x;                    // B * N * (SS/128)
    const int SB = SS / 128;                 // 16
    int sb = blk % SB;  blk /= SB;
    int n  = blk % NHD; blk /= NHD;
    int b  = blk;
    const int s_base = sb * 128 + wave * 16; // this wave owns 16 query rows
    const int s      = s_base + col;         // lane's softmax row

    const bf16_t* qrow  = X  + ((size_t)b * SS + s) * DD;
    const bf16_t* Kbase = Kp + (((size_t)b * NHD + n) * SS) * DD;
    const bf16_t* Vbase = Vt + (((size_t)b * NHD + n) * DD) * (size_t)SS;
    const int*    mrow  = mask + ((size_t)b * SS + s) * (size_t)SS;

    const unsigned ksBase = (unsigned)(uintptr_t)&Ks[0][0];
    const unsigned vsBase = (unsigned)(uintptr_t)&Vs[0][0];

    // stage one (K,V) f-tile into LDS buffer `bufn`: 256 threads x 128B
    auto prefetch = [&](int f0, int bufn) {
        const char* kg = (const char*)(Kbase + (size_t)f0 * DD) + tid * 64;
        unsigned    kl = ksBase + (unsigned)bufn * (32 * DD * 2) + tid * 64;
        #pragma unroll
        for (int i = 0; i < 4; ++i) async_cp16(kl + i * 16, kg + i * 16);
        const char* vg = (const char*)(Vbase + (size_t)tid * SS + f0);
        unsigned    vl = vsBase + (unsigned)bufn * (DD * 32 * 2) + tid * 64;
        #pragma unroll
        for (int i = 0; i < 4; ++i) async_cp16(vl + i * 16, vg + i * 16);
    };

    // hoist Q B-operand (f-invariant) into registers: 8 x v16bf
    BV16 qreg[8];
    #pragma unroll
    for (int dq = 0; dq < 8; ++dq) {
        qreg[dq].q[0] = *(const uint4*)(qrow + dq * 32 + hi * 8);
        qreg[dq].q[1] = *(const uint4*)(qrow + dq * 32 + hi * 8 + 16);
    }

    float m_run = -3.0e38f, l_run = 0.0f;
    v8f o[16];
    #pragma unroll
    for (int t = 0; t < 16; ++t) o[t] = v8f_zero();

    prefetch(0, 0);
    int buf = 0;
    for (int f0 = 0; f0 < SS; f0 += 32) {
        wait_async0();
        __syncthreads();     // tile `buf` ready; all waves done reading buf^1
        if (f0 + 32 < SS) prefetch(f0 + 32, buf ^ 1);

        // ---- S^T tiles from LDS: c0 = K[f0..f0+15].Q^T, c1 = K[f0+16..31].Q^T
        const bf16_t* kt = &Ks[buf][0];
        v8f c0 = v8f_zero(), c1 = v8f_zero();
        #pragma unroll
        for (int dq = 0; dq < 8; ++dq) {
            const int d0 = dq * 32;
            BV16 a0, a1;
            a0.q[0] = *(const uint4*)(kt + col * DD + d0 + hi * 8);
            a0.q[1] = *(const uint4*)(kt + col * DD + d0 + hi * 8 + 16);
            a1.q[0] = *(const uint4*)(kt + (16 + col) * DD + d0 + hi * 8);
            a1.q[1] = *(const uint4*)(kt + (16 + col) * DD + d0 + hi * 8 + 16);
            c0 = wmma_bf16(a0.v, qreg[dq].v, c0);
            c1 = wmma_bf16(a1.v, qreg[dq].v, c1);
        }
        // lane holds 16 score values for its row s: f = f0 + hi*8 + i (i<8)
        //                                           f = f0 + 16 + hi*8 + (i-8)
        float sc[16];
        #pragma unroll
        for (int i = 0; i < 8; ++i) { sc[i] = c0[i]; sc[8 + i] = c1[i]; }

        const int* mp = mrow + f0 + hi * 8;
        int4 m0 = *(const int4*)(mp);
        int4 m1 = *(const int4*)(mp + 4);
        int4 m2 = *(const int4*)(mp + 16);
        int4 m3 = *(const int4*)(mp + 20);
        int mv[16] = {m0.x, m0.y, m0.z, m0.w, m1.x, m1.y, m1.z, m1.w,
                      m2.x, m2.y, m2.z, m2.w, m3.x, m3.y, m3.z, m3.w};
        #pragma unroll
        for (int i = 0; i < 16; ++i)
            sc[i] += (1.0f - (float)mv[i]) * -10000.0f;

        // ---- online softmax over f (per lane row s; partner lane is lane^16)
        float tmax = sc[0];
        #pragma unroll
        for (int i = 1; i < 16; ++i) tmax = fmaxf(tmax, sc[i]);
        tmax = fmaxf(tmax, __shfl_xor(tmax, 16, 32));
        const float m_new = fmaxf(m_run, tmax);
        const float alpha = __expf(m_run - m_new);

        float psum = 0.0f;
        BV16 pa;   // exp(scores) packed directly into P.V A-operand layout
        #pragma unroll
        for (int i = 0; i < 16; ++i) {
            float p = __expf(sc[i] - m_new);
            psum += p;
            pa.h[i] = f2bf(p);
        }
        psum += __shfl_xor(psum, 16, 32);
        l_run = l_run * alpha + psum;
        m_run = m_new;

        // alpha lives in lane = s; O rows are s = r + 8*hi -> redistribute
        float ar[8];
        #pragma unroll
        for (int r = 0; r < 8; ++r) ar[r] = __shfl(alpha, hi * 8 + r, 32);

        // ---- O += P(16x32) x V(32 x 256), 16 d-chunks, V from LDS
        const bf16_t* vt0 = &Vs[buf][0];
        #pragma unroll
        for (int t = 0; t < 16; ++t) {
            v8f ot = o[t];
            #pragma unroll
            for (int r = 0; r < 8; ++r) ot[r] *= ar[r];
            const bf16_t* vrow = vt0 + (t * 16 + col) * 32 + hi * 8;
            BV16 bv;
            bv.q[0] = *(const uint4*)(vrow);
            bv.q[1] = *(const uint4*)(vrow + 16);
            o[t] = wmma_bf16(pa.v, bv.v, ot);
        }
        buf ^= 1;
    }

    // normalize rows by 1/l and accumulate over heads
    const float linv = 1.0f / l_run;
    float lr[8];
    #pragma unroll
    for (int r = 0; r < 8; ++r) lr[r] = __shfl(linv, hi * 8 + r, 32);
    #pragma unroll
    for (int t = 0; t < 16; ++t) {
        #pragma unroll
        for (int r = 0; r < 8; ++r) {
            float v = o[t][r] * lr[r];
            atomicAdd(out + ((size_t)b * SS + s_base + hi * 8 + r) * DD + t * 16 + col, v);
        }
    }
}

// ---------------- launcher ----------------
extern "C" void kernel_launch(void* const* d_in, const int* in_sizes, int n_in,
                              void* d_out, int out_size, void* d_ws, size_t ws_size,
                              hipStream_t stream) {
    const float* input = (const float*)d_in[0];
    const int*   mask  = (const int*)d_in[1];
    const float* qk_w  = (const float*)d_in[2];
    const float* qk_b  = (const float*)d_in[3];
    const float* v_w   = (const float*)d_in[4];
    const float* v_b   = (const float*)d_in[5];
    float* out = (float*)d_out;

    char* ws = (char*)d_ws;
    size_t off = 0;
    bf16_t* Xbf = (bf16_t*)(ws + off); off += (size_t)BB * SS * DD * 2;        // 4 MiB
    bf16_t* Wqt = (bf16_t*)(ws + off); off += (size_t)FF * DD * 2;             // 1.5 MiB
    bf16_t* Wvt = (bf16_t*)(ws + off); off += (size_t)FF * DD * 2;             // 1.5 MiB
    bf16_t* Kp  = (bf16_t*)(ws + off); off += (size_t)BB * NHD * SS * DD * 2;  // 48 MiB
    bf16_t* Vt  = (bf16_t*)(ws + off); off += (size_t)BB * NHD * SS * DD * 2;  // 48 MiB
    (void)ws_size; (void)in_sizes; (void)n_in; (void)out_size;

    const int nX = BB * SS * DD;                 // 2,097,152
    zero_f32<<<(nX + 255) / 256, 256, 0, stream>>>(out, nX);
    cvt_bf16<<<(nX + 255) / 256, 256, 0, stream>>>(input, Xbf, nX);
    const int nW = FF * DD;                      // 786,432
    transpose_w<<<(nW + 255) / 256, 256, 0, stream>>>(qk_w, Wqt);
    transpose_w<<<(nW + 255) / 256, 256, 0, stream>>>(v_w, Wvt);

    const int projGrid = BB * (SS / 16) * (FF / 16);   // 98304 tiles, 1 wave each
    proj_kernel<0><<<projGrid, 32, 0, stream>>>(Xbf, Wqt, qk_b, Kp);
    proj_kernel<1><<<projGrid, 32, 0, stream>>>(Xbf, Wvt, v_b, Vt);

    attn_kernel<<<BB * NHD * (SS / 128), 256, 0, stream>>>(Xbf, Kp, Vt, mask, out);
}
